// EnhancedLSTMfcnKan_89953795047921
// MI455X (gfx1250) — compile-verified, matrix operand order checked by reading, not run on hardware
//
#include <hip/hip_runtime.h>
#include <hip/hip_bf16.h>
#include <math.h>

typedef __attribute__((ext_vector_type(16))) _Float16 v16h;
typedef __attribute__((ext_vector_type(8)))  float    v8f;

// ---------------------------------------------------------------------------
// Fast scalar ops (gfx1250 hardware RCP/TANH transcendentals)
// ---------------------------------------------------------------------------
__device__ __forceinline__ float fast_rcp(float x) {
#if __has_builtin(__builtin_amdgcn_rcpf)
  return __builtin_amdgcn_rcpf(x);
#else
  return 1.f / x;
#endif
}
__device__ __forceinline__ float fast_tanh(float x) {
#if __has_builtin(__builtin_amdgcn_tanhf)
  return __builtin_amdgcn_tanhf(x);
#else
  return tanhf(x);
#endif
}
__device__ __forceinline__ float sigm(float x) { return fast_rcp(1.f + __expf(-x)); }

// ---------------------------------------------------------------------------
// WMMA fragment helpers (CDNA5 wave32 layouts, CDNA5 ISA §7.12.2)
//   A 16x32 f16 : lanes 0-15 row M=lane, K=0..7/16..23 ; lanes 16-31 K=8..15/24..31
//   B 32x16 f16 : mirrored (N = lane&15, same K grouping)
//   C/D 16x16 f32: VGPR r -> M = r + (lane>=16 ? 8:0), N = lane&15
// All operand layouts below are arranged K-contiguous per lane so the
// compiler vectorizes fragment loads into ds_load_b128 / global_load_b128.
// ---------------------------------------------------------------------------
__device__ __forceinline__ int lane_id() { return threadIdx.x & 31; }

template <typename F>
__device__ __forceinline__ v16h make_a_frag(F get /* (m,k)->float */) {
  v16h a;
  const int l = lane_id(), m = l & 15, kb = (l >> 4) << 3;
#pragma unroll
  for (int i = 0; i < 8; ++i) a[i] = (_Float16)get(m, kb + i);
#pragma unroll
  for (int i = 0; i < 8; ++i) a[8 + i] = (_Float16)get(m, kb + 16 + i);
  return a;
}
template <typename F>
__device__ __forceinline__ v16h make_a_frag_h(F get /* (m,k)->_Float16 */) {
  v16h a;
  const int l = lane_id(), m = l & 15, kb = (l >> 4) << 3;
#pragma unroll
  for (int i = 0; i < 8; ++i) a[i] = get(m, kb + i);
#pragma unroll
  for (int i = 0; i < 8; ++i) a[8 + i] = get(m, kb + 16 + i);
  return a;
}
template <typename F>
__device__ __forceinline__ v16h make_b_frag(F get /* (k,n)->float */) {
  v16h b;
  const int l = lane_id(), n = l & 15, kb = (l >> 4) << 3;
#pragma unroll
  for (int i = 0; i < 8; ++i) b[i] = (_Float16)get(kb + i, n);
#pragma unroll
  for (int i = 0; i < 8; ++i) b[8 + i] = (_Float16)get(kb + 16 + i, n);
  return b;
}
template <typename F>
__device__ __forceinline__ v16h make_b_frag_h(F get /* (k,n)->_Float16 */) {
  v16h b;
  const int l = lane_id(), n = l & 15, kb = (l >> 4) << 3;
#pragma unroll
  for (int i = 0; i < 8; ++i) b[i] = get(kb + i, n);
#pragma unroll
  for (int i = 0; i < 8; ++i) b[8 + i] = get(kb + 16 + i, n);
  return b;
}
template <typename F>
__device__ __forceinline__ void store_c_frag(const v8f& c, F put /* (m,n,v) */) {
  const int l = lane_id(), n = l & 15, mb = (l >> 4) << 3;
#pragma unroll
  for (int r = 0; r < 8; ++r) put(mb + r, n, c[r]);
}
__device__ __forceinline__ v8f wmma_f16(v16h a, v16h b, v8f c) {
  return __builtin_amdgcn_wmma_f32_16x16x32_f16(false, a, false, b, (short)0, c,
                                                false, false);
}

// ---------------------------------------------------------------------------
// Kernel 1: KAN-LSTM. Persistent single workgroup (8 waves). Weights staged
// once in LDS *transposed to N-major* (K contiguous per lane -> ds_load_b128
// B-fragments); x_t staged per step; h maintained as f16. Hardware rcp/tanh
// on the serial gate path.
// ---------------------------------------------------------------------------
__global__ void k_lstm(const float* __restrict__ x,    // (32,1024,64)
                       const float* __restrict__ w1,   // (192,16)
                       const float* __restrict__ b1,   // (16)
                       const float* __restrict__ w2,   // (16,512)
                       const float* __restrict__ b2,   // (512)
                       float* __restrict__ hidden)     // (32,1024,128)
{
  extern __shared__ float smem[];
  float* sc    = smem;                  // 32*128 cell state (f32)
  float* sg1   = sc + 4096;             // 32*16 raw GEMM1 out
  float* sgate = sg1 + 512;             // 32*512 raw GEMM2 out
  float* b1s   = sgate + 16384;         // 16
  float* b2s   = b1s + 16;              // 512
  _Float16* sh16 = (_Float16*)(b2s + 512);            // 32*128 h (f16)
  _Float16* sx16 = (_Float16*)((float*)sh16 + 2048);  // 32*64 x_t (f16)
  _Float16* sg1h = (_Float16*)((float*)sx16 + 1024);  // 32*16 silu(g1) (f16)
  _Float16* w1t  = (_Float16*)((float*)sg1h + 256);   // 16 x 192 (n-major!)
  _Float16* w2t  = (_Float16*)((float*)w1t + 1536);   // 512 x 16 (n-major!)

  const int tid = threadIdx.x, wid = tid >> 5;

  for (int i = tid; i < 32 * 128; i += 256) { sc[i] = 0.f; sh16[i] = (_Float16)0.f; }
  for (int i = tid; i < 16 * 192; i += 256) {      // transpose w1 -> [n][k]
    const int n = i / 192, k = i % 192;
    w1t[i] = (_Float16)w1[k * 16 + n];
  }
  for (int i = tid; i < 512 * 16; i += 256) {      // transpose w2 -> [n][k]
    const int n = i >> 4, k = i & 15;
    w2t[i] = (_Float16)w2[k * 512 + n];
  }
  if (tid < 16) b1s[tid] = b1[tid];
  for (int i = tid; i < 512; i += 256) b2s[i] = b2[i];
  __syncthreads();

  for (int t = 0; t < 1024; ++t) {
    for (int i = tid; i < 32 * 64; i += 256) {     // stage x_t (32x64) as f16
      const int row = i >> 6, kk = i & 63;
      sx16[i] = (_Float16)x[((size_t)row * 1024 + t) * 64 + kk];
    }
    __syncthreads();
    // GEMM1: concat(x_t, h)(32x192) @ w1(192x16) ; waves 0,1 take the M-tiles
    if (wid < 2) {
      const int m0 = wid * 16;
      v8f acc = {};
      for (int kc = 0; kc < 6; ++kc) {
        const int k0 = kc * 32;
        v16h a = make_a_frag_h([&](int m, int k) -> _Float16 {
          const int row = m0 + m, kk = k0 + k;
          return (kk < 64) ? sx16[row * 64 + kk] : sh16[row * 128 + (kk - 64)];
        });
        v16h b = make_b_frag_h([&](int k, int n) -> _Float16 {
          return w1t[n * 192 + k0 + k];            // K-contiguous -> b128
        });
        acc = wmma_f16(a, b, acc);
      }
      store_c_frag(acc, [&](int m, int n, float v) { sg1[(m0 + m) * 16 + n] = v; });
    }
    __syncthreads();
    for (int i = tid; i < 32 * 16; i += 256) {     // bias + SiLU -> f16
      const float v = sg1[i] + b1s[i & 15];
      sg1h[i] = (_Float16)(v * sigm(v));
    }
    __syncthreads();
    // GEMM2: (32x16 pad32) @ w2(16x512): 64 tile jobs, 8 per wave
    for (int j = 0; j < 8; ++j) {
      const int job = wid * 8 + j;
      const int m0 = (job >> 5) * 16, n0 = (job & 31) * 16;
      v16h a = make_a_frag_h([&](int m, int k) -> _Float16 {
        return (k < 16) ? sg1h[(m0 + m) * 16 + k] : (_Float16)0.f;
      });
      v16h b = make_b_frag_h([&](int k, int n) -> _Float16 {
        return (k < 16) ? w2t[(n0 + n) * 16 + k] : (_Float16)0.f;  // contiguous
      });
      v8f acc = {};
      acc = wmma_f16(a, b, acc);
      store_c_frag(acc, [&](int m, int n, float v) { sgate[(m0 + m) * 512 + n0 + n] = v; });
    }
    __syncthreads();
    for (int i = tid; i < 32 * 128; i += 256) {    // gates (hw rcp/tanh)
      const int b = i >> 7, n = i & 127;
      const float gi = sgate[b * 512 + n]       + b2s[n];
      const float gf = sgate[b * 512 + 128 + n] + b2s[128 + n];
      const float gg = sgate[b * 512 + 256 + n] + b2s[256 + n];
      const float go = sgate[b * 512 + 384 + n] + b2s[384 + n];
      const float cn = sigm(gf) * sc[i] + sigm(gi) * fast_tanh(gg);
      const float hv = sigm(go) * fast_tanh(cn);
      sc[i] = cn; sh16[i] = (_Float16)hv;
      hidden[((size_t)b * 1024 + t) * 128 + n] = hv;
    }
    __syncthreads();
  }
}

// ---------------------------------------------------------------------------
// Kernel 2: row-normalize hidden -> hn (f16)
// ---------------------------------------------------------------------------
__global__ void k_hn(const float* __restrict__ hidden, _Float16* __restrict__ hn16) {
  const int row = blockIdx.x;          // b*1024 + t
  const int tid = threadIdx.x;         // 128 threads
  __shared__ float red[128];
  const float v = hidden[(size_t)row * 128 + tid];
  red[tid] = v * v;
  __syncthreads();
  for (int s = 64; s > 0; s >>= 1) { if (tid < s) red[tid] += red[tid + s]; __syncthreads(); }
  const float nrm = fmaxf(sqrtf(red[0]), 1e-8f);
  hn16[(size_t)row * 128 + tid] = (_Float16)(v / nrm);
}

__device__ __forceinline__ float wave_incl_scan(float v, int lane) {
#pragma unroll
  for (int off = 1; off < 32; off <<= 1) {
    const float u = __shfl_up(v, off, 32);
    if (lane >= off) v += u;
  }
  return v;
}

// ---------------------------------------------------------------------------
// Kernel 3: clustering. 1 block/batch. 16-row strips of S via WMMA; both A and
// B fragments read hn16 K-contiguously from global (vectorized b128).
// Wave32 prefix scan + running column accumulator => O(T) cumsum state.
// ---------------------------------------------------------------------------
__global__ void k_cluster(const _Float16* __restrict__ hn16,
                          const float* __restrict__ hidden,
                          float* __restrict__ lastb,   // (32,128)
                          float* __restrict__ corrb)   // (32,128)
{
  extern __shared__ float smem[];
  float* Srow    = smem;               // 16*1024
  float* colAcc  = Srow + 16 * 1024;   // 1024
  float* diagA   = colAcc + 1024;      // 1024
  float* rowTot  = diagA + 1024;       // 1024
  float* prefTot = rowTot + 1024;      // 1024
  float* redv    = prefTot + 1024;     // 256
  int*   redi    = (int*)(redv + 256); // 256

  const int b = blockIdx.x;
  const int tid = threadIdx.x, wid = tid >> 5, lane = tid & 31;
  const _Float16* hb = hn16 + (size_t)b * 1024 * 128;

  for (int i = tid; i < 1024; i += 256) colAcc[i] = 0.f;
  __syncthreads();

  for (int rt = 0; rt < 64; ++rt) {
    const int i0 = rt * 16;
    for (int j = 0; j < 8; ++j) {          // 64 column tiles / 8 waves
      const int j0 = (wid * 8 + j) * 16;
      v8f acc = {};
      for (int kc = 0; kc < 4; ++kc) {
        const int k0 = kc * 32;
        v16h a = make_a_frag_h([&](int m, int k) -> _Float16 {
          return hb[(i0 + m) * 128 + k0 + k];
        });
        v16h bf = make_b_frag_h([&](int k, int n) -> _Float16 {  // hn^T
          return hb[(j0 + n) * 128 + k0 + k];
        });
        acc = wmma_f16(a, bf, acc);
      }
      store_c_frag(acc, [&](int m, int n, float v) {
        const int gi = i0 + m, gj = j0 + n;
        float s = __expf(-5.5f * (1.f - v));
        if (gi == gj) s = 0.f;
        Srow[m * 1024 + gj] = s;
      });
    }
    __syncthreads();
    for (int rr = 0; rr < 2; ++rr) {       // per-row prefix along j
      const int r = wid * 2 + rr;
      float carry = 0.f;
      for (int c0 = 0; c0 < 1024; c0 += 32) {
        float v = Srow[r * 1024 + c0 + lane];
        v = wave_incl_scan(v, lane) + carry;
        Srow[r * 1024 + c0 + lane] = v;
        carry = __shfl(v, 31, 32);
      }
    }
    __syncthreads();
    for (int r = 0; r < 16; ++r) {         // fold rows in order, record diag
      for (int kk = tid; kk < 1024; kk += 256) colAcc[kk] += Srow[r * 1024 + kk];
      __syncthreads();
      if (tid == 0) {
        const int gi = i0 + r;
        diagA[gi]  = colAcc[gi];
        rowTot[gi] = Srow[r * 1024 + 1023];
      }
      __syncthreads();
    }
  }
  if (wid == 0) {                          // prefix of row totals
    float carry = 0.f;
    for (int c0 = 0; c0 < 1024; c0 += 32) {
      float v = rowTot[c0 + lane];
      v = wave_incl_scan(v, lane) + carry;
      prefTot[c0 + lane] = v;
      carry = __shfl(v, 31, 32);
    }
  }
  __syncthreads();
  const float full = prefTot[1023];
  float best = -3.4e38f; int bestI = 1;
  for (int i = 1 + tid; i < 1024; i += 256) {
    const float TL = diagA[i - 1];
    const float TR = prefTot[i - 1] - TL;
    const float BL = colAcc[i - 1] - TL;
    const float BR = full - TL - TR - BL;
    const float ii = (float)i, nn = (float)(1024 - i);
    const float d = TL / (ii * ii) + BR / (nn * nn) - (TR + BL) / (ii * nn);
    if (d > best) { best = d; bestI = i; }
  }
  redv[tid] = best; redi[tid] = bestI;
  __syncthreads();
  for (int s = 128; s > 0; s >>= 1) {
    if (tid < s) {
      if (redv[tid + s] > redv[tid] ||
          (redv[tid + s] == redv[tid] && redi[tid + s] < redi[tid])) {
        redv[tid] = redv[tid + s]; redi[tid] = redi[tid + s];
      }
    }
    __syncthreads();
  }
  const int cut = redi[0];
  const float inv = fast_rcp((float)cut);
  for (int hh = tid; hh < 128; hh += 256) {
    lastb[b * 128 + hh] = hidden[((size_t)b * 1024 + 1023) * 128 + hh];
    float s = 0.f;
    for (int t = 0; t < cut; ++t) s += hidden[((size_t)b * 1024 + t) * 128 + hh];
    corrb[b * 128 + hh] = s * inv;
  }
}

// ---------------------------------------------------------------------------
// Kernel 4: VAE heads. One block, WMMA for the four GEMMs.
// ---------------------------------------------------------------------------
__global__ void k_heads(const float* __restrict__ corr, const float* __restrict__ last,
                        const float* __restrict__ eps_z, const float* __restrict__ eps_cat,
                        const float* __restrict__ mu_w, const float* __restrict__ mu_b,
                        const float* __restrict__ st_w, const float* __restrict__ st_b,
                        const float* __restrict__ muc_w, const float* __restrict__ muc_b,
                        const float* __restrict__ stc_w, const float* __restrict__ stc_b,
                        float* __restrict__ o_mu, float* __restrict__ o_std,
                        float* __restrict__ o_muc, float* __restrict__ o_stdc,
                        float* __restrict__ x1)
{
  extern __shared__ float smem[];
  float* smu = smem;            // 32*128
  float* sst = smu + 4096;      // 32*128
  float* sxz = sst + 4096;      // 32*256
  const int tid = threadIdx.x, wid = tid >> 5;

  for (int j = 0; j < 4; ++j) {   // corr(32x128) @ {mu_w,std_w}(128x128)
    const int job = wid * 4 + j;
    const int mat = job >> 4, mt = (job >> 3) & 1, nt = job & 7;
    const float* W = mat ? st_w : mu_w;
    float* O = mat ? sst : smu;
    const int m0 = mt * 16, n0 = nt * 16;
    v8f acc = {};
    for (int kc = 0; kc < 4; ++kc) {
      const int k0 = kc * 32;
      v16h a = make_a_frag([&](int m, int k) -> float { return corr[(m0 + m) * 128 + k0 + k]; });
      v16h b = make_b_frag([&](int k, int n) -> float { return W[(k0 + k) * 128 + n0 + n]; });
      acc = wmma_f16(a, b, acc);
    }
    store_c_frag(acc, [&](int m, int n, float v) { O[(m0 + m) * 128 + n0 + n] = v; });
  }
  __syncthreads();
  for (int i = tid; i < 4096; i += 256) {
    const int n = i & 127;
    const float mu = smu[i] + mu_b[n];
    const float sv = sst[i] + st_b[n];
    const float sp = (sv > 20.f) ? sv : log1pf(__expf(sv));
    o_mu[i] = mu; o_std[i] = sp;
    const float z = mu + sp * eps_z[i];
    const int b = i >> 7;
    sxz[b * 256 + n] = last[i];
    sxz[b * 256 + 128 + n] = z;
  }
  __syncthreads();
  for (int j = 0; j < 4; ++j) {   // xz(32x256) @ {muc_w,stdc_w}(256x128)
    const int job = wid * 4 + j;
    const int mat = job >> 4, mt = (job >> 3) & 1, nt = job & 7;
    const float* W = mat ? stc_w : muc_w;
    float* O = mat ? sst : smu;
    const int m0 = mt * 16, n0 = nt * 16;
    v8f acc = {};
    for (int kc = 0; kc < 8; ++kc) {
      const int k0 = kc * 32;
      v16h a = make_a_frag([&](int m, int k) -> float { return sxz[(m0 + m) * 256 + k0 + k]; });
      v16h b = make_b_frag([&](int k, int n) -> float { return W[(k0 + k) * 128 + n0 + n]; });
      acc = wmma_f16(a, b, acc);
    }
    store_c_frag(acc, [&](int m, int n, float v) { O[(m0 + m) * 128 + n0 + n] = v; });
  }
  __syncthreads();
  for (int i = tid; i < 4096; i += 256) {
    const int n = i & 127;
    const float mc = smu[i] + muc_b[n];
    const float sv = sst[i] + stc_b[n];
    const float sp = (sv > 20.f) ? sv : log1pf(__expf(sv));
    o_muc[i] = mc; o_stdc[i] = sp;
    x1[i] = mc + sp * eps_cat[i];
  }
}

// ---------------------------------------------------------------------------
// Weight prep: f32 (Cout,Cin,Kw) -> f16 tap-major [(dk*Cout+co)*Cin+ci]
// so conv A-fragments are K(=ci)-contiguous f16 -> global_load_b128.
// ---------------------------------------------------------------------------
__global__ void k_wprep(const float* __restrict__ w, _Float16* __restrict__ wf,
                        int Cout, int Cin, int Kw) {
  const int i = blockIdx.x * 256 + threadIdx.x;
  if (i >= Cout * Cin * Kw) return;
  const int dk = i / (Cout * Cin);
  const int r  = i % (Cout * Cin);
  const int co = r / Cin, ci = r % Cin;
  wf[i] = (_Float16)w[((size_t)co * Cin + ci) * Kw + dk];
}

// ---------------------------------------------------------------------------
// Kernel 5: causal conv1d as implicit GEMM with LDS tile reuse.
// One block = one (batch, 16-wide time tile). Input patch staged ONCE in LDS
// as f16 *time-major* [tt][Cin] (pad + SE scale folded), so B-fragments are
// ci-contiguous -> ds_load_b128. Weights come from the prepped f16 tensor.
// in_layout 0: x-style (b,t,ci) ; 1: (b,ci,t)
// ---------------------------------------------------------------------------
__global__ void k_conv(const float* __restrict__ in, const _Float16* __restrict__ wf,
                       const float* __restrict__ bias, const float* __restrict__ scale,
                       float* __restrict__ out, int Cin, int Cout, int Kw, int in_layout)
{
  extern __shared__ float smem[];
  _Float16* sxin = (_Float16*)smem;      // [padT][Cin]
  const int tid = threadIdx.x, wid = tid >> 5;
  const int padT = 16 + Kw - 1;
  const int b = blockIdx.x >> 6;         // 64 time tiles per batch
  const int t0 = (blockIdx.x & 63) * 16;

  if (in_layout == 1 && t0 + 16 < 1024)  // speculative prefetch of next tile
    __builtin_prefetch(&in[((size_t)b * Cin) * 1024 + t0 + 16], 0, 1);

  if (in_layout == 0) {
    // idx = tt*Cin + ci: ci contiguous -> coalesced global, linear LDS store
    for (int idx = tid; idx < Cin * padT; idx += 256) {
      const int tt = idx / Cin, ci = idx % Cin;
      const int t = t0 + tt - (Kw - 1);
      float v = (t >= 0) ? in[((size_t)b * 1024 + t) * 64 + ci] : 0.f;
      sxin[idx] = (_Float16)v;
    }
  } else {
    // idx = ci*padT + tt: tt contiguous -> coalesced global, scatter LDS store
    for (int idx = tid; idx < Cin * padT; idx += 256) {
      const int ci = idx / padT, tt = idx % padT;
      const int t = t0 + tt - (Kw - 1);
      float v = (t >= 0) ? in[((size_t)b * Cin + ci) * 1024 + t] : 0.f;
      if (scale) v *= scale[b * Cin + ci];
      sxin[tt * Cin + ci] = (_Float16)v;
    }
  }
  __syncthreads();

  const int coTiles = Cout >> 4, kChunks = Cin >> 5;
  for (int job = wid; job < coTiles; job += 8) {
    const int c0 = job * 16;
    v8f acc = {};
    for (int dk = 0; dk < Kw; ++dk) {
      const _Float16* wtap = wf + (size_t)dk * Cout * Cin;
      for (int kc = 0; kc < kChunks; ++kc) {
        const int ci0 = kc * 32;
        v16h a = make_a_frag_h([&](int m, int k) -> _Float16 {
          return wtap[(size_t)(c0 + m) * Cin + ci0 + k];    // contiguous f16
        });
        v16h bf = make_b_frag_h([&](int k, int n) -> _Float16 {
          return sxin[(n + dk) * Cin + ci0 + k];            // contiguous f16
        });
        acc = wmma_f16(a, bf, acc);
      }
    }
    store_c_frag(acc, [&](int m, int n, float v) {
      out[((size_t)b * Cout + c0 + m) * 1024 + t0 + n] = v + bias[c0 + m];
    });
  }
}

// ---------------------------------------------------------------------------
// BN stats / BN+GELU / time-mean / SE MLP / final FC + log_softmax
// ---------------------------------------------------------------------------
__global__ void k_bnstats(const float* __restrict__ y, float* __restrict__ mean,
                          float* __restrict__ var, int C) {
  const int c = blockIdx.x, tid = threadIdx.x;
  __shared__ float s1[256], s2[256];
  float a = 0.f, b = 0.f;
  for (int i = tid; i < 32 * 1024; i += 256) {
    const int bb = i >> 10, t = i & 1023;
    const float v = y[((size_t)bb * C + c) * 1024 + t];
    a += v; b += v * v;
  }
  s1[tid] = a; s2[tid] = b; __syncthreads();
  for (int s = 128; s > 0; s >>= 1) {
    if (tid < s) { s1[tid] += s1[tid + s]; s2[tid] += s2[tid + s]; }
    __syncthreads();
  }
  if (tid == 0) {
    const float m = s1[0] / 32768.f;
    mean[c] = m; var[c] = s2[0] / 32768.f - m * m;
  }
}

__global__ void k_bngelu(float* __restrict__ y, const float* __restrict__ mean,
                         const float* __restrict__ var, const float* __restrict__ g,
                         const float* __restrict__ bta, int C) {
  const size_t i = (size_t)blockIdx.x * 256 + threadIdx.x;
  const int c = (int)((i >> 10) % C);
  float v = y[i];
  v = (v - mean[c]) * rsqrtf(var[c] + 1e-3f) * g[c] + bta[c];
  y[i] = 0.5f * v * (1.f + erff(v * 0.70710678f));   // exact GELU
}

__global__ void k_rowmean(const float* __restrict__ y, float* __restrict__ out) {
  const int row = blockIdx.x;          // b*C + c
  __shared__ float s[256];
  float a = 0.f;
  for (int t = threadIdx.x; t < 1024; t += 256) a += y[(size_t)row * 1024 + t];
  s[threadIdx.x] = a; __syncthreads();
  for (int ss = 128; ss > 0; ss >>= 1) {
    if (threadIdx.x < ss) s[threadIdx.x] += s[threadIdx.x + ss];
    __syncthreads();
  }
  if (threadIdx.x == 0) out[row] = s[0] / 1024.f;
}

__global__ void k_se(const float* __restrict__ ymean, const float* __restrict__ w1,
                     const float* __restrict__ w2, float* __restrict__ sout,
                     int C, int Hid) {
  __shared__ float t1[32 * 16];
  const int tid = threadIdx.x;
  for (int i = tid; i < 32 * Hid; i += 256) {
    const int b = i / Hid, j = i % Hid;
    float a = 0.f;
    for (int ci = 0; ci < C; ++ci) a += ymean[b * C + ci] * w1[ci * Hid + j];
    t1[b * Hid + j] = fmaxf(a, 0.f);
  }
  __syncthreads();
  for (int i = tid; i < 32 * C; i += 256) {
    const int b = i / C, cc = i % C;
    float a = 0.f;
    for (int j = 0; j < Hid; ++j) a += t1[b * Hid + j] * w2[j * C + cc];
    sout[i] = sigm(a);
  }
}

__global__ void k_fc(const float* __restrict__ x1, const float* __restrict__ xc,
                     const float* __restrict__ fw, const float* __restrict__ fb,
                     float* __restrict__ out) {
  __shared__ float lg[32 * 10];
  const int tid = threadIdx.x;
  for (int i = tid; i < 320; i += 256) {
    const int b = i / 10, n = i % 10;
    float a = fb[n];
    for (int k = 0; k < 128; ++k) a += x1[b * 128 + k] * fw[k * 10 + n];
    for (int k = 0; k < 128; ++k) a += xc[b * 128 + k] * fw[(128 + k) * 10 + n];
    lg[i] = a;
  }
  __syncthreads();
  if (tid < 32) {
    float mx = -3.4e38f;
    for (int n = 0; n < 10; ++n) mx = fmaxf(mx, lg[tid * 10 + n]);
    float se = 0.f;
    for (int n = 0; n < 10; ++n) se += __expf(lg[tid * 10 + n] - mx);
    const float l = mx + __logf(se);
    for (int n = 0; n < 10; ++n) out[tid * 10 + n] = lg[tid * 10 + n] - l;
  }
}

// ---------------------------------------------------------------------------
extern "C" void kernel_launch(void* const* d_in, const int* in_sizes, int n_in,
                              void* d_out, int out_size, void* d_ws, size_t ws_size,
                              hipStream_t stream) {
  const float* x        = (const float*)d_in[0];
  const float* eps_z    = (const float*)d_in[1];
  const float* eps_cat  = (const float*)d_in[2];
  const float* kan_w1   = (const float*)d_in[3];
  const float* kan_b1   = (const float*)d_in[4];
  const float* kan_w2   = (const float*)d_in[5];
  const float* kan_b2   = (const float*)d_in[6];
  const float* conv1_w  = (const float*)d_in[7];
  const float* conv1_b  = (const float*)d_in[8];
  const float* conv2_w  = (const float*)d_in[9];
  const float* conv2_b  = (const float*)d_in[10];
  const float* conv3_w  = (const float*)d_in[11];
  const float* conv3_b  = (const float*)d_in[12];
  const float* bn1_g    = (const float*)d_in[13];
  const float* bn1_b    = (const float*)d_in[14];
  const float* bn2_g    = (const float*)d_in[15];
  const float* bn2_b    = (const float*)d_in[16];
  const float* bn3_g    = (const float*)d_in[17];
  const float* bn3_b    = (const float*)d_in[18];
  const float* se1_w1   = (const float*)d_in[19];
  const float* se1_w2   = (const float*)d_in[20];
  const float* se2_w1   = (const float*)d_in[21];
  const float* se2_w2   = (const float*)d_in[22];
  const float* fc_mu_w  = (const float*)d_in[23];
  const float* fc_mu_b  = (const float*)d_in[24];
  const float* fc_st_w  = (const float*)d_in[25];
  const float* fc_st_b  = (const float*)d_in[26];
  const float* fc_muc_w = (const float*)d_in[27];
  const float* fc_muc_b = (const float*)d_in[28];
  const float* fc_stc_w = (const float*)d_in[29];
  const float* fc_stc_b = (const float*)d_in[30];
  const float* fc_w     = (const float*)d_in[31];
  const float* fc_b     = (const float*)d_in[32];

  float* out = (float*)d_out;   // [logits 320][mu 4096][std 4096][mu_cat 4096][std_cat 4096]
  float* o_logits = out;
  float* o_mu     = out + 320;
  float* o_std    = out + 320 + 4096;
  float* o_muc    = out + 320 + 8192;
  float* o_stdc   = out + 320 + 12288;

  // workspace layout (floats)
  float* ws = (float*)d_ws;
  size_t off = 0;
  float*     hidden = ws + off; off += (size_t)32 * 1024 * 128;
  _Float16*  hn16   = (_Float16*)(ws + off); off += (size_t)32 * 1024 * 128 / 2;
  float*     y1     = ws + off; off += (size_t)32 * 128 * 1024;
  float*     y2     = ws + off; off += (size_t)32 * 256 * 1024;
  float*     y3     = ws + off; off += (size_t)32 * 128 * 1024;
  float*     lastb  = ws + off; off += 4096;
  float*     corrb  = ws + off; off += 4096;
  float*     x1     = ws + off; off += 4096;
  float*     m1 = ws + off; off += 128;  float* v1 = ws + off; off += 128;
  float*     m2 = ws + off; off += 256;  float* v2 = ws + off; off += 256;
  float*     m3 = ws + off; off += 128;  float* v3 = ws + off; off += 128;
  float*     ymean = ws + off; off += 8192;
  float*     s1 = ws + off; off += 4096;
  float*     s2 = ws + off; off += 8192;
  float*     xc = ws + off; off += 4096;
  _Float16*  wf1 = (_Float16*)(ws + off); off += (size_t)128 * 64 * 8 / 2;    // 65536 h
  _Float16*  wf2 = (_Float16*)(ws + off); off += (size_t)256 * 128 * 5 / 2;   // 163840 h
  _Float16*  wf3 = (_Float16*)(ws + off); off += (size_t)128 * 256 * 3 / 2;   // 98304 h

  // ---- conv weight prep (f16 tap-major, runs up front) ----
  k_wprep<<<(128 * 64 * 8 + 255) / 256, 256, 0, stream>>>(conv1_w, wf1, 128, 64, 8);
  k_wprep<<<(256 * 128 * 5 + 255) / 256, 256, 0, stream>>>(conv2_w, wf2, 256, 128, 5);
  k_wprep<<<(128 * 256 * 3 + 255) / 256, 256, 0, stream>>>(conv3_w, wf3, 128, 256, 3);

  // ---- LSTM (persistent single workgroup; ~119KB dynamic LDS, f16 operands) ----
  const size_t lstm_smem =
      (size_t)(4096 + 512 + 16384 + 16 + 512          // f32 regions
               + 2048 + 1024 + 256 + 1536 + 4096)     // f16 regions (float slots)
      * sizeof(float);
  k_lstm<<<1, 256, lstm_smem, stream>>>(x, kan_w1, kan_b1, kan_w2, kan_b2, hidden);

  // ---- normalize for gram ----
  k_hn<<<32 * 1024, 128, 0, stream>>>(hidden, hn16);

  // ---- clustering: 1 block / batch ----
  const size_t cl_smem = (size_t)(16 * 1024 + 5 * 1024 + 256) * sizeof(float) + 256 * sizeof(int);
  k_cluster<<<32, 256, cl_smem, stream>>>(hn16, hidden, lastb, corrb);

  // ---- VAE heads ----
  const size_t hd_smem = (size_t)(4096 + 4096 + 8192) * sizeof(float);
  k_heads<<<1, 256, hd_smem, stream>>>(corrb, lastb, eps_z, eps_cat,
                                       fc_mu_w, fc_mu_b, fc_st_w, fc_st_b,
                                       fc_muc_w, fc_muc_b, fc_stc_w, fc_stc_b,
                                       o_mu, o_std, o_muc, o_stdc, x1);

  // ---- conv stack (2048 blocks = 32 batches x 64 time tiles each) ----
  k_conv<<<2048, 256, (size_t)23 * 64 * 2, stream>>>(x, wf1, conv1_b, nullptr, y1,
                                                     64, 128, 8, 0);
  k_bnstats<<<128, 256, 0, stream>>>(y1, m1, v1, 128);
  k_bngelu<<<32 * 128 * 1024 / 256, 256, 0, stream>>>(y1, m1, v1, bn1_g, bn1_b, 128);
  k_rowmean<<<32 * 128, 256, 0, stream>>>(y1, ymean);
  k_se<<<1, 256, 0, stream>>>(ymean, se1_w1, se1_w2, s1, 128, 8);

  k_conv<<<2048, 256, (size_t)20 * 128 * 2, stream>>>(y1, wf2, conv2_b, s1, y2,
                                                      128, 256, 5, 1);
  k_bnstats<<<256, 256, 0, stream>>>(y2, m2, v2, 256);
  k_bngelu<<<32 * 256 * 1024 / 256, 256, 0, stream>>>(y2, m2, v2, bn2_g, bn2_b, 256);
  k_rowmean<<<32 * 256, 256, 0, stream>>>(y2, ymean);
  k_se<<<1, 256, 0, stream>>>(ymean, se2_w1, se2_w2, s2, 256, 16);

  k_conv<<<2048, 256, (size_t)18 * 256 * 2, stream>>>(y2, wf3, conv3_b, s2, y3,
                                                      256, 128, 3, 1);
  k_bnstats<<<128, 256, 0, stream>>>(y3, m3, v3, 128);
  k_bngelu<<<32 * 128 * 1024 / 256, 256, 0, stream>>>(y3, m3, v3, bn3_g, bn3_b, 128);
  k_rowmean<<<32 * 128, 256, 0, stream>>>(y3, xc);

  // ---- final FC + log_softmax ----
  k_fc<<<1, 256, 0, stream>>>(x1, xc, fc_w, fc_b, o_logits);
}
